// PaperLSTMClassifier_67800353735262
// MI455X (gfx1250) — compile-verified, matrix operand order checked by reading, not run on hardware
//
#include <hip/hip_runtime.h>
#include <hip/hip_bf16.h>

typedef __attribute__((ext_vector_type(16))) _Float16 v16h;
typedef __attribute__((ext_vector_type(8)))  float    v8f;

#define LAYERS 5
#define HDIM   64
#define CATDIM 128
#define NGATES 4

// Branch-free activations: lower to v_exp_f32 / v_rcp_f32 TRANS ops only.
__device__ __forceinline__ float fast_sigmoid(float x) {
    return __builtin_amdgcn_rcpf(1.0f + __expf(-x));
}
__device__ __forceinline__ float fast_tanh(float x) {
    // 1 - 2/(e^{2x}+1); exp->inf gives 1, exp->0 gives -1. No branches, no NaN.
    return 1.0f - 2.0f * __builtin_amdgcn_rcpf(__expf(2.0f * x) + 1.0f);
}

// Offset of element (row m [A] or col n [B], local k'' in 0..31) inside a
// 512-half 16x32 fragment tile, per CDNA5 16-bit A-matrix layout.
__device__ __forceinline__ int frag_off(int mn, int kk) {
    int lane = mn + 16 * ((kk >> 3) & 1);
    int half = (kk & 7) + 8 * (kk >> 4);
    return lane * 16 + half;
}

// ---------------------------------------------------------------------------
// Prep: repack fp32 weights [L,H,CAT] x 4 gates -> f16 WMMA B-fragments.
// ---------------------------------------------------------------------------
__global__ void prep_weights_kernel(const float* __restrict__ Wi,
                                    const float* __restrict__ Wf,
                                    const float* __restrict__ Wg,
                                    const float* __restrict__ Wo,
                                    _Float16* __restrict__ out) {
    int idx = blockIdx.x * blockDim.x + threadIdx.x;
    const int total = LAYERS * NGATES * HDIM * CATDIM;
    if (idx >= total) return;
    int k = idx & (CATDIM - 1);
    int n = (idx >> 7) & (HDIM - 1);
    int g = (idx >> 13) & 3;
    int l = idx >> 15;
    const float* W = (g == 0) ? Wi : (g == 1) ? Wf : (g == 2) ? Wg : Wo;
    float v = W[(l * HDIM + n) * CATDIM + k];
    int colgrp = n >> 4, nn = n & 15;
    int kt = k >> 5, kk = k & 31;
    int tile = ((l * NGATES + g) * 4 + colgrp) * 4 + kt;
    out[tile * 512 + frag_off(nn, kk)] = (_Float16)v;
}

// ---------------------------------------------------------------------------
// Prep: f16 embedding rows in A-fragment half order so each lane's fragment
// halves are contiguous: within each 32-dim group, order [0..7,16..23,8..15,
// 24..31]; lane-half hi reads 16 halves at offset hi*16 (one v16h load).
// ---------------------------------------------------------------------------
__global__ void prep_emb_kernel(const float* __restrict__ emb, int VH,
                                _Float16* __restrict__ out) {
    int idx = blockIdx.x * blockDim.x + threadIdx.x;
    if (idx >= VH) return;
    int kk  = idx & (HDIM - 1);
    int row = idx >> 6;
    int b3 = (kk >> 3) & 1, b4 = (kk >> 4) & 1;
    int pos = (kk & 7) + 8 * (b3 * 2 + b4) + 32 * (kk >> 5);
    out[row * HDIM + pos] = (_Float16)emb[idx];
}

// ---------------------------------------------------------------------------
// Bitonic sort (descending by length) of row indices: groups rows of similar
// length into the same 16-row tile so each workgroup only runs the recurrence
// to the max length of its tile (~2x total work reduction for uniform lens).
// ---------------------------------------------------------------------------
__global__ void sort_rows_kernel(const int* __restrict__ lengths, int B,
                                 int* __restrict__ perm) {
    __shared__ int key[1024];
    __shared__ int val[1024];
    const int tid = threadIdx.x;
    key[tid] = (tid < B) ? lengths[tid] : -1;
    val[tid] = tid;
    __syncthreads();
    for (int k = 2; k <= 1024; k <<= 1) {
        for (int j = k >> 1; j > 0; j >>= 1) {
            int ixj = tid ^ j;
            if (ixj > tid) {
                bool descBlock = ((tid & k) == 0);
                int k0 = key[tid], k1 = key[ixj];
                bool doswap = descBlock ? (k0 < k1) : (k0 > k1);
                if (doswap) {
                    int v0 = val[tid], v1 = val[ixj];
                    key[tid] = k1; key[ixj] = k0;
                    val[tid] = v1; val[ixj] = v0;
                }
            }
            __syncthreads();
        }
    }
    if (tid < B) perm[tid] = val[tid];
}

// ---------------------------------------------------------------------------
// Persistent LSTM kernel. One workgroup = 16 (length-sorted) batch rows,
// 4 waves; wave w owns output columns 16w..16w+15 of every gate/layer; all
// weight fragments (5L x 4g x 4kt = 640 VGPRs) are register-resident.
// ---------------------------------------------------------------------------
__launch_bounds__(128, 1)
__global__ void lstm_persistent_kernel(
    const int*   __restrict__ x, const int* __restrict__ lengths,
    const _Float16* __restrict__ emb16,
    const float* __restrict__ b_i, const float* __restrict__ b_f,
    const float* __restrict__ b_g, const float* __restrict__ b_o,
    const float* __restrict__ init_h,
    const float* __restrict__ head_w, const float* __restrict__ head_b,
    const _Float16* __restrict__ Wfrag, const int* __restrict__ perm,
    float* __restrict__ out, int Bsz, int Tsz)
{
    __shared__ _Float16 hbuf[2][LAYERS][1024]; // h A-frags (K tiles 2,3), dbl-buf
    __shared__ _Float16 cur[2][1024];          // inter-layer input A-frags
    __shared__ float    outacc[16];
    __shared__ int      smax;

    const int tid   = threadIdx.x;
    const int wave  = tid >> 5;
    const int lane  = tid & 31;
    const int hi    = lane >> 4;
    const int lan16 = lane & 15;
    const int col   = wave * 16 + lan16;
    const int r0    = blockIdx.x * 16;

    const int tsel    = (col >> 5) & 1;
    const int halfoff = (col & 7) + 8 * ((col >> 4) & 1);
    const int colb3   = ((col >> 3) & 1) * 16;

    // -------- register-resident weights --------
    v16h wreg[LAYERS][NGATES][4];
    #pragma unroll
    for (int l = 0; l < LAYERS; ++l)
        #pragma unroll
        for (int g = 0; g < NGATES; ++g)
            #pragma unroll
            for (int kt = 0; kt < 4; ++kt)
                wreg[l][g][kt] = *(const v16h*)(
                    Wfrag + (((l * NGATES + g) * 4 + wave) * 4 + kt) * 512 + lane * 16);

    // -------- persistent per-lane state --------
    float bias[LAYERS][NGATES];
    float h_state[LAYERS][8];
    float c_state[LAYERS][8];
    #pragma unroll
    for (int l = 0; l < LAYERS; ++l) {
        bias[l][0] = b_i[l * HDIM + col];
        bias[l][1] = b_f[l * HDIM + col];
        bias[l][2] = b_g[l * HDIM + col];
        bias[l][3] = b_o[l * HDIM + col];
        float h0 = fast_tanh(init_h[l * HDIM + col]);
        #pragma unroll
        for (int v = 0; v < 8; ++v) { h_state[l][v] = h0; c_state[l][v] = 0.0f; }
    }
    int lenv[8];
    int lmax = 0;
    #pragma unroll
    for (int v = 0; v < 8; ++v) {
        lenv[v] = lengths[perm[r0 + v + 8 * hi]];
        lmax = max(lmax, lenv[v]);
    }
    const int rowA = perm[r0 + lan16];     // row for this lane's A-frag gather
    const float hw = head_w[col];

    if (tid == 0) smax = 0;
    __syncthreads();
    atomicMax(&smax, lmax);

    // seed h fragments for t=0 (buffer 0)
    #pragma unroll
    for (int l = 0; l < LAYERS; ++l) {
        #pragma unroll
        for (int v = 0; v < 8; ++v) {
            int off = tsel * 512 + ((v + 8 * hi) + colb3) * 16 + halfoff;
            hbuf[0][l][off] = (_Float16)h_state[l][v];
        }
    }
    if (tid < 16) outacc[tid] = 0.0f;
    __syncthreads();
    const int tmax = smax;                 // max length in this 16-row tile

    // -------- serial recurrence, only to the tile's max length --------
    for (int t = 0; t < tmax; ++t) {
        const int rb = t & 1, wb = rb ^ 1;

        // freeze masks, hoisted out of the layer loop (depend on t,row only)
        float am[8];
        #pragma unroll
        for (int v = 0; v < 8; ++v)
            am[v] = (t < lenv[v]) ? 1.0f : 0.0f;

        // layer-0 input fragments: two v16h loads from f16 frag-ordered emb
        const int tok = x[rowA * Tsz + t];
        const _Float16* erow = emb16 + tok * HDIM;
        v16h ax0 = *(const v16h*)(erow + hi * 16);
        v16h ax1 = *(const v16h*)(erow + 32 + hi * 16);

        #pragma unroll
        for (int l = 0; l < LAYERS; ++l) {
            v16h a0, a1;
            if (l == 0) { a0 = ax0; a1 = ax1; }
            else {
                const _Float16* src = cur[(l - 1) & 1];
                a0 = *(const v16h*)&src[lane * 16];
                a1 = *(const v16h*)&src[512 + lane * 16];
            }
            const _Float16* hsrc = hbuf[rb][l];
            v16h ah0 = *(const v16h*)&hsrc[lane * 16];
            v16h ah1 = *(const v16h*)&hsrc[512 + lane * 16];

            float gate[NGATES][8];
            #pragma unroll
            for (int g = 0; g < NGATES; ++g) {
                v8f acc;
                #pragma unroll
                for (int v = 0; v < 8; ++v) acc[v] = bias[l][g];
                acc = __builtin_amdgcn_wmma_f32_16x16x32_f16(false, a0,  false, wreg[l][g][0], (short)0, acc, false, false);
                acc = __builtin_amdgcn_wmma_f32_16x16x32_f16(false, a1,  false, wreg[l][g][1], (short)0, acc, false, false);
                acc = __builtin_amdgcn_wmma_f32_16x16x32_f16(false, ah0, false, wreg[l][g][2], (short)0, acc, false, false);
                acc = __builtin_amdgcn_wmma_f32_16x16x32_f16(false, ah1, false, wreg[l][g][3], (short)0, acc, false, false);
                #pragma unroll
                for (int v = 0; v < 8; ++v) gate[g][v] = acc[v];
            }

            #pragma unroll
            for (int v = 0; v < 8; ++v) {
                float ig = fast_sigmoid(gate[0][v]);
                float fg = fast_sigmoid(gate[1][v]);
                float gg = fast_tanh(gate[2][v]);
                float og = fast_sigmoid(gate[3][v]);
                float cn = fg * c_state[l][v] + ig * gg;
                float hn = og * fast_tanh(cn);
                // branch-free freeze: old + am*(new-old), am in {0,1}
                float hl = h_state[l][v] + am[v] * (hn - h_state[l][v]);
                float cl = c_state[l][v] + am[v] * (cn - c_state[l][v]);
                h_state[l][v] = hl;
                c_state[l][v] = cl;
                int off = tsel * 512 + ((v + 8 * hi) + colb3) * 16 + halfoff;
                _Float16 hh = (_Float16)hl;
                hbuf[wb][l][off] = hh;                    // recurrent input @ t+1
                if (l < LAYERS - 1) cur[l & 1][off] = hh; // next layer input @ t
            }
            __syncthreads();
        }
    }

    // -------- head: logits[row] = sum_col h_L[m][col]*head_w[col] + head_b --
    #pragma unroll
    for (int v = 0; v < 8; ++v)
        atomicAdd(&outacc[v + 8 * hi], h_state[LAYERS - 1][v] * hw);
    __syncthreads();
    if (tid < 16) out[perm[r0 + tid]] = outacc[tid] + head_b[0];
}

extern "C" void kernel_launch(void* const* d_in, const int* in_sizes, int n_in,
                              void* d_out, int out_size, void* d_ws, size_t ws_size,
                              hipStream_t stream) {
    const int*   x       = (const int*)d_in[0];
    const int*   lengths = (const int*)d_in[1];
    const float* emb     = (const float*)d_in[2];
    const float* W_i     = (const float*)d_in[3];
    const float* W_f     = (const float*)d_in[4];
    const float* W_g     = (const float*)d_in[5];
    const float* W_o     = (const float*)d_in[6];
    const float* b_i     = (const float*)d_in[7];
    const float* b_f     = (const float*)d_in[8];
    const float* b_g     = (const float*)d_in[9];
    const float* b_o     = (const float*)d_in[10];
    const float* init_h  = (const float*)d_in[11];
    const float* head_w  = (const float*)d_in[12];
    const float* head_b  = (const float*)d_in[13];

    const int B  = in_sizes[1];         // lengths has B elements
    const int T  = in_sizes[0] / B;     // x has B*T elements
    const int VH = in_sizes[2];         // emb has V*H elements

    const size_t wbytes = (size_t)LAYERS * NGATES * HDIM * CATDIM * sizeof(_Float16);
    _Float16* Wfrag = (_Float16*)d_ws;
    int*      perm  = (int*)((char*)d_ws + wbytes);
    _Float16* emb16 = (_Float16*)((char*)d_ws + wbytes + 4096);

    const int total = LAYERS * NGATES * HDIM * CATDIM;
    prep_weights_kernel<<<(total + 255) / 256, 256, 0, stream>>>(W_i, W_f, W_g, W_o, Wfrag);
    prep_emb_kernel<<<(VH + 255) / 256, 256, 0, stream>>>(emb, VH, emb16);
    sort_rows_kernel<<<1, 1024, 0, stream>>>(lengths, B, perm);

    lstm_persistent_kernel<<<B / 16, 128, 0, stream>>>(
        x, lengths, emb16, b_i, b_f, b_g, b_o, init_h, head_w, head_b,
        Wfrag, perm, (float*)d_out, B, T);
}
